// GroupQueryAttention_37529424233062
// MI455X (gfx1250) — matrix operand order repaired and checked by reading, not compiled, not run on hardware
//
#include <hip/hip_runtime.h>
#include <hip/hip_bf16.h>

// GQA forward for gfx1250 (MI455X): all matmuls via v_wmma_f32_16x16x32_f16.
// H=16, G=4, D=2048, HD=128, KV=512, bs=2, s=2048.

typedef __attribute__((ext_vector_type(16))) _Float16 v16h;
typedef __attribute__((ext_vector_type(8)))  _Float16 v8h;
typedef __attribute__((ext_vector_type(8)))  float    v8f;

static __device__ __forceinline__ v8f wmma16(v16h a, v16h b, v8f c) {
    // D = A(16x32,f16) * B(32x16,f16) + C(16x16,f32)
    return __builtin_amdgcn_wmma_f32_16x16x32_f16(false, a, false, b, (short)0, c, false, false);
}

static __device__ __forceinline__ v16h mk16(v8h lo, v8h hi) {
    v16h r;
#pragma unroll
    for (int i = 0; i < 8; ++i) { r[i] = lo[i]; r[8 + i] = hi[i]; }
    return r;
}

static __device__ __forceinline__ float half_max(float x) {
#pragma unroll
    for (int m = 1; m <= 8; m <<= 1) x = fmaxf(x, __shfl_xor(x, m, 32));
    return x;   // reduction over the 16 lanes sharing bit4 (rows stay lane-local)
}
static __device__ __forceinline__ float half_sum(float x) {
#pragma unroll
    for (int m = 1; m <= 8; m <<= 1) x += __shfl_xor(x, m, 32);
    return x;
}

#define GPITCH 40   // LDS row pitch in halves (80 B, keeps 16B alignment)

// ---------------------------------------------------------------------------
// Tiled GEMM:  C[M,N] = A[M,K] @ B[K,N] + bias[N]
// A: float or _Float16 row-major; B: float row-major (converted to f16 in LDS);
// C: f16 (OUT_F16) or f32.  Requires M%128==0, N%128==0, K%32==0.
// Block = 256 threads (8 waves); block tile 128x128; K step 32.
// Wave (wm = w&3, wn = w>>2) computes a 32x64 slice: 2 A-frags x 4 B-frags
// = 8 WMMAs per K-step.  AI ~= 32 FLOP/B of HBM traffic.
// ---------------------------------------------------------------------------
template <typename TA, bool OUT_F16>
__global__ __launch_bounds__(256)
void gemm_bias_kernel(const TA* __restrict__ A, const float* __restrict__ B,
                      const float* __restrict__ bias, void* __restrict__ Cout,
                      int M, int N, int K)
{
    __shared__ _Float16 As[128 * GPITCH];
    __shared__ _Float16 Bt[128 * GPITCH];   // transposed: Bt[n][k]

    const int tid  = threadIdx.x;
    const int wave = tid >> 5;
    const int lane = tid & 31;
    const int hi   = lane >> 4;
    const int ln   = lane & 15;
    const int wm   = wave & 3;      // 4 row groups of 32
    const int wn   = wave >> 2;     // 2 col groups of 64
    const int mBase = blockIdx.y * 128;
    const int nBase = blockIdx.x * 128;

    const int aRow = tid >> 1;           // 0..127
    const int aK   = (tid & 1) * 16;     // 0,16
    const int bK   = tid >> 3;           // 0..31
    const int bN   = (tid & 7) * 16;     // 0..112

    v8f acc[2][4];
#pragma unroll
    for (int i = 0; i < 2; ++i)
#pragma unroll
        for (int nb = 0; nb < 4; ++nb) acc[i][nb] = v8f{};

    const TA*    srcA = A + (size_t)(mBase + aRow) * K + aK;
    const float* srcB = B + (size_t)bK * N + nBase + bN;

    for (int kb = 0; kb < K; kb += 32) {
        {   // stage A tile (convert to f16)
            _Float16* dst = &As[aRow * GPITCH + aK];
#pragma unroll
            for (int i = 0; i < 16; ++i) dst[i] = (_Float16)srcA[kb + i];
        }
        {   // stage B tile transposed (coalesced global read, strided LDS write)
            const float* sb = srcB + (size_t)kb * N;
#pragma unroll
            for (int i = 0; i < 16; ++i) Bt[(bN + i) * GPITCH + bK] = (_Float16)sb[i];
        }
        if (kb + 32 < K) {   // pull next K-slab toward L0/L2 across the barrier
            __builtin_prefetch(srcA + kb + 32, 0, 0);
            __builtin_prefetch(srcB + (size_t)(kb + 32) * N, 0, 0);
        }
        __syncthreads();

        // A fragments: m = ln, k = (e>>3)*16 + hi*8 + (e&7)
        v16h af[2];
#pragma unroll
        for (int i = 0; i < 2; ++i) {
            const _Float16* ap = &As[(wm * 32 + i * 16 + ln) * GPITCH];
            af[i] = mk16(*(const v8h*)(ap + hi * 8), *(const v8h*)(ap + 16 + hi * 8));
        }
#pragma unroll
        for (int nb = 0; nb < 4; ++nb) {
            // B fragment: n = ln, k = (e>>3)*16 + hi*8 + (e&7)
            const _Float16* bp = &Bt[(wn * 64 + nb * 16 + ln) * GPITCH];
            v16h bf = mk16(*(const v8h*)(bp + hi * 8), *(const v8h*)(bp + 16 + hi * 8));
            acc[0][nb] = wmma16(af[0], bf, acc[0][nb]);
            acc[1][nb] = wmma16(af[1], bf, acc[1][nb]);
        }
        __syncthreads();
    }

#pragma unroll
    for (int i = 0; i < 2; ++i) {
#pragma unroll
        for (int nb = 0; nb < 4; ++nb) {
#pragma unroll
            for (int r = 0; r < 8; ++r) {
                const int row = mBase + wm * 32 + i * 16 + hi * 8 + r;
                const int col = nBase + wn * 64 + nb * 16 + ln;
                const float v = acc[i][nb][r] + bias[col];
                if (OUT_F16) ((_Float16*)Cout)[(size_t)row * N + col] = (_Float16)v;
                else         ((float*)   Cout)[(size_t)row * N + col] = v;
            }
        }
    }
}

// ---------------------------------------------------------------------------
// Flash attention with the reference's raw-reshape KV expansion:
//   k_exp[b,h,s2,d] = Kproj[b*2048 + h*128 + s2/16][(s2%4)*128 + d]
// Grid: (32 q-blocks of 64, 16 heads, 2 batches); block 128 = 4 waves,
// each wave owns 16 query rows. 32 keys per step; trip count block-uniform.
// ---------------------------------------------------------------------------
__global__ __launch_bounds__(128)
void gqa_attn_kernel(const _Float16* __restrict__ Qp, const _Float16* __restrict__ Kp,
                     const _Float16* __restrict__ Vp, _Float16* __restrict__ Op)
{
    __shared__ _Float16 Pl[4 * 16 * GPITCH];   // per-wave P staging (C->A layout)

    const int tid  = threadIdx.x;
    const int wave = tid >> 5;
    const int lane = tid & 31;
    const int hi   = lane >> 4;
    const int ln   = lane & 15;
    const int h    = blockIdx.y;
    const int b    = blockIdx.z;
    const int qb   = blockIdx.x * 64 + wave * 16;
    const float NEGINF = -__builtin_inff();
    const float scale  = 0.08838834764831845f;   // 1/sqrt(128)

    // Preload Q fragments (A layout, 4 chunks of k=32 over d=128)
    v16h qf[4];
    {
        const _Float16* qrow = Qp + ((size_t)(b * 2048 + qb + ln)) * 2048 + h * 128;
#pragma unroll
        for (int c = 0; c < 4; ++c)
            qf[c] = mk16(*(const v8h*)(qrow + c * 32 + hi * 16),
                         *(const v8h*)(qrow + c * 32 + hi * 16 + 8));
    }

    v8f oacc[8];
#pragma unroll
    for (int cd = 0; cd < 8; ++cd) oacc[cd] = v8f{};
    float m_run[8], l_run[8];
#pragma unroll
    for (int r = 0; r < 8; ++r) { m_run[r] = NEGINF; l_run[r] = 0.0f; }

    const size_t headKV = ((size_t)(b * 2048 + h * 128)) * 512;  // Kproj row base * 512
    _Float16* Pw = &Pl[wave * 16 * GPITCH];
    const int steps = 2 * (blockIdx.x + 1);      // uniform across all 4 waves

    for (int j = 0; j < steps; ++j) {
        // ---- S = Q @ K^T for 32 keys (two 16-key tiles t=2j, 2j+1) ----
        v8f s0 = v8f{}, s1 = v8f{};
#pragma unroll
        for (int c = 0; c < 4; ++c) {
            const _Float16* k0 = Kp + headKV + (size_t)(2 * j) * 512
                                 + (ln & 3) * 128 + c * 32 + hi * 16;
            v16h kf0 = *(const v16h*)k0;          // 32B aligned
            v16h kf1 = *(const v16h*)(k0 + 512);  // next tile row
            s0 = wmma16(qf[c], kf0, s0);
            s1 = wmma16(qf[c], kf1, s1);
        }

        // ---- online softmax (rows are lane-local: row = hi*8 + r) ----
        float p0[8], p1[8];
#pragma unroll
        for (int r = 0; r < 8; ++r) {
            const int q = qb + hi * 8 + r;
            float a0 = s0[r] * scale; if (32 * j + ln      > q) a0 = NEGINF;
            float a1 = s1[r] * scale; if (32 * j + 16 + ln > q) a1 = NEGINF;
            float rm   = half_max(fmaxf(a0, a1));
            float mnew = fmaxf(m_run[r], rm);
            float alpha = __expf(m_run[r] - mnew);
            p0[r] = __expf(a0 - mnew);
            p1[r] = __expf(a1 - mnew);
            float rs = half_sum(p0[r] + p1[r]);
            l_run[r] = l_run[r] * alpha + rs;
            m_run[r] = mnew;
#pragma unroll
            for (int cd = 0; cd < 8; ++cd) oacc[cd][r] *= alpha;
        }

        // ---- P: C layout -> A layout via per-wave LDS round trip ----
        __syncthreads();
#pragma unroll
        for (int r = 0; r < 8; ++r) {
            Pw[(hi * 8 + r) * GPITCH + ln]      = (_Float16)p0[r];
            Pw[(hi * 8 + r) * GPITCH + 16 + ln] = (_Float16)p1[r];
        }
        __syncthreads();
        const _Float16* pp = &Pw[ln * GPITCH];
        v16h pf = mk16(*(const v8h*)(pp + hi * 16), *(const v8h*)(pp + hi * 16 + 8));

        // ---- O += P @ V  (V gathered via the raw-reshape mapping) ----
#pragma unroll
        for (int cd = 0; cd < 8; ++cd) {
            // B frag element e: key s2 = 32j + hi*16 + e -> Vproj row (2j+hi),
            // group g = e&3, col = g*128 + cd*16 + ln. Only 4 distinct values.
            const _Float16* v0 = Vp + headKV + (size_t)(2 * j + hi) * 512 + cd * 16 + ln;
            _Float16 vv[4];
#pragma unroll
            for (int g = 0; g < 4; ++g) vv[g] = v0[g * 128];
            v16h vf;
#pragma unroll
            for (int e = 0; e < 16; ++e) vf[e] = vv[e & 3];
            oacc[cd] = wmma16(pf, vf, oacc[cd]);
        }
    }

    // ---- normalize and write attn output [b, q, h*128 + d] as f16 ----
#pragma unroll
    for (int r = 0; r < 8; ++r) {
        const float inv = 1.0f / l_run[r];
        const size_t orow = ((size_t)(b * 2048 + qb + hi * 8 + r)) * 2048 + h * 128;
#pragma unroll
        for (int cd = 0; cd < 8; ++cd)
            Op[orow + cd * 16 + ln] = (_Float16)(oacc[cd][r] * inv);
    }
}

// ---------------------------------------------------------------------------
extern "C" void kernel_launch(void* const* d_in, const int* in_sizes, int n_in,
                              void* d_out, int out_size, void* d_ws, size_t ws_size,
                              hipStream_t stream)
{
    const float* X  = (const float*)d_in[0];
    const float* Wq = (const float*)d_in[1];
    const float* bq = (const float*)d_in[2];
    const float* Wk = (const float*)d_in[3];
    const float* bk = (const float*)d_in[4];
    const float* Wv = (const float*)d_in[5];
    const float* bv = (const float*)d_in[6];
    const float* Wo = (const float*)d_in[7];
    const float* bo = (const float*)d_in[8];
    float* out = (float*)d_out;

    // workspace: Qp 16MB | Kp 4MB | Vp 4MB | Op 16MB  (total ~42MB)
    char* ws = (char*)d_ws;
    _Float16* Qp = (_Float16*)(ws);
    _Float16* Kp = (_Float16*)(ws + (size_t)16777216);
    _Float16* Vp = (_Float16*)(ws + (size_t)16777216 + 4194304);
    _Float16* Op = (_Float16*)(ws + (size_t)16777216 + 2 * 4194304);

    const dim3 gblk(256);
    const dim3 ablk(128);
    // Projections (fp32 in -> f16 out, fp32 accumulate)
    gemm_bias_kernel<float, true><<<dim3(2048 / 128, 4096 / 128), gblk, 0, stream>>>(
        X, Wq, bq, (void*)Qp, 4096, 2048, 2048);
    gemm_bias_kernel<float, true><<<dim3(512 / 128, 4096 / 128), gblk, 0, stream>>>(
        X, Wk, bk, (void*)Kp, 4096, 512, 2048);
    gemm_bias_kernel<float, true><<<dim3(512 / 128, 4096 / 128), gblk, 0, stream>>>(
        X, Wv, bv, (void*)Vp, 4096, 512, 2048);
    // Causal flash attention with raw-reshape KV expansion
    gqa_attn_kernel<<<dim3(32, 16, 2), ablk, 0, stream>>>(Qp, Kp, Vp, Op);
    // Output projection (f16 in -> fp32 out)
    gemm_bias_kernel<_Float16, false><<<dim3(2048 / 128, 4096 / 128), gblk, 0, stream>>>(
        Op, Wo, bo, (void*)out, 4096, 2048, 2048);
}
